// MultiheadAttention_35424890257587
// MI455X (gfx1250) — matrix operand order repaired and checked by reading, not compiled, not run on hardware
//
#include <hip/hip_runtime.h>
#include <hip/hip_bf16.h>

// MultiheadAttention forward for MI455X (gfx1250, wave32, WMMA).
// GEMM(qkv, bf16 out) -> flash attention (bf16 in/out) -> GEMM(out proj, f32 out).
// Matmuls: v_wmma_f32_16x16x32_bf16.
// Data movers, double-buffered to overlap with WMMA compute:
//   - TDM tensor_load_to_lds for the attention K tile (TENSORcnt)
//   - global_load_async_to_lds_b128 for GEMM2's bf16 A tile (ASYNCcnt)

typedef __attribute__((ext_vector_type(8)))  __bf16 v8bf_t;
typedef __attribute__((ext_vector_type(16))) __bf16 v16bf_t;
typedef __attribute__((ext_vector_type(8)))  float  v8f_t;
typedef __attribute__((ext_vector_type(4)))  int    v4i_t;
typedef __attribute__((ext_vector_type(8)))  int    v8i_t;

__device__ __forceinline__ v8bf_t cvt8v(float4 a, float4 b) {
    v8bf_t r;
    r[0] = (__bf16)a.x; r[1] = (__bf16)a.y; r[2] = (__bf16)a.z; r[3] = (__bf16)a.w;
    r[4] = (__bf16)b.x; r[5] = (__bf16)b.y; r[6] = (__bf16)b.z; r[7] = (__bf16)b.w;
    return r;
}

__device__ __forceinline__ v8bf_t ld8(const __bf16* p) { return *(const v8bf_t*)p; }

__device__ __forceinline__ v16bf_t combine8(v8bf_t lo, v8bf_t hi) {
    v16bf_t r;
#pragma unroll
    for (int i = 0; i < 8; ++i) { r[i] = lo[i]; r[i + 8] = hi[i]; }
    return r;
}

__device__ __forceinline__ v8f_t wmma_bf16(v16bf_t a, v16bf_t b, v8f_t c) {
    return __builtin_amdgcn_wmma_f32_16x16x32_bf16(false, a, false, b, (short)0, c,
                                                   false, false);
}

__device__ __forceinline__ unsigned lds_addr(const void* p) {
    return (unsigned)(unsigned long long)p;
}

// Async 16-byte global -> LDS copy (per lane), ASYNCcnt-tracked.
__device__ __forceinline__ void async_cp16(unsigned ldsaddr, const __bf16* g) {
    asm volatile("global_load_async_to_lds_b128 %0, %1, off"
                 :: "v"(ldsaddr), "v"(g)
                 : "memory");
}

__device__ __forceinline__ void async_wait0() {
    asm volatile("s_wait_asynccnt 0" ::: "memory");
}

// TDM: DMA a 2D bf16 tile (tile_w x tile_h, row stride `stride` elements)
// from global memory into LDS. Issue under a wave-uniform branch (EXEC is
// ignored by TDM). Tracked with TENSORcnt.
__device__ __forceinline__ void tdm_load_tile_bf16(unsigned lds, const __bf16* g,
                                                   int tile_w, int tile_h,
                                                   int stride) {
    const unsigned long long ga = (unsigned long long)g;
    v4i_t g0;
    g0[0] = 1;                                                // count=1, user desc
    g0[1] = (int)lds;                                         // lds_addr[31:0]
    g0[2] = (int)(unsigned)ga;                                // global_addr[31:0]
    g0[3] = (int)((unsigned)((ga >> 32) & 0x01FFFFFFu) |      // global_addr[56:32]
                  0x80000000u);                               // type=2 ("image")
    v8i_t g1;
    g1[0] = 0x00010000;          // workgroup_mask=0, data_size=1 (2 bytes)
    g1[1] = tile_w << 16;        // tensor_dim0[15:0]  (bits 63:48)
    g1[2] = tile_h << 16;        // tensor_dim0[31:16]=0, tensor_dim1[15:0]
    g1[3] = tile_w << 16;        // tensor_dim1[31:16]=0, tile_dim0
    g1[4] = tile_h & 0xFFFF;     // tile_dim1, tile_dim2=0
    g1[5] = stride;              // tensor_dim0_stride[31:0]
    g1[6] = 0;
    g1[7] = 0;
    asm volatile("tensor_load_to_lds %0, %1" :: "s"(g0), "s"(g1) : "memory");
}

__device__ __forceinline__ void tensor_wait0() {
    __builtin_amdgcn_s_wait_tensorcnt(0);
}

// ---------------------------------------------------------------------------
// C(M,N) = A(M,K) * W(N,K)^T + bias(N)
// AT: f32 (convert on stage) or bf16 (async-LDS copy on stage). OT: f32/bf16.
// 256 threads = 8 waves; tile 128(M) x 128(N); wave tile 32x64; K-step 32;
// double-buffered LDS, one barrier per K-step, loads overlap WMMA.
// ---------------------------------------------------------------------------
template <typename AT, typename OT>
__global__ __launch_bounds__(256) void gemm_bias_kernel(
    const AT* __restrict__ A, const float* __restrict__ W,
    const float* __restrict__ bias, OT* __restrict__ C,
    int M, int N, int K) {
    __shared__ alignas(16) __bf16 sA[2][128 * 32];
    __shared__ alignas(16) __bf16 sW[2][128 * 32];

    const int t    = threadIdx.x;
    const int lane = t & 31;
    const int wid  = t >> 5;
    const int half = lane >> 4;
    const int ln   = lane & 15;
    const int wrow = wid & 3;   // 4 wave-rows of 32
    const int wcol = wid >> 2;  // 2 wave-cols of 64
    const int m0   = blockIdx.y * 128;
    const int n0   = blockIdx.x * 128;

    v8f_t acc[2][4] = {};

    const int sr = t >> 1;        // staging row (0..127)
    const int sk = (t & 1) * 16;  // staging K offset (0 or 16)

    // ---- prologue: stage tile k0=0 into buffer 0 ----
    {
        if constexpr (sizeof(AT) == 4) {
            const float4* as = (const float4*)(A + (size_t)(m0 + sr) * K + sk);
            *(v8bf_t*)&sA[0][sr * 32 + sk]     = cvt8v(as[0], as[1]);
            *(v8bf_t*)&sA[0][sr * 32 + sk + 8] = cvt8v(as[2], as[3]);
        } else {
            const __bf16* as = (const __bf16*)A + (size_t)(m0 + sr) * K + sk;
            const unsigned l = lds_addr(&sA[0][sr * 32 + sk]);
            async_cp16(l, as);
            async_cp16(l + 16, as + 8);
        }
        const float4* ws = (const float4*)(W + (size_t)(n0 + sr) * K + sk);
        *(v8bf_t*)&sW[0][sr * 32 + sk]     = cvt8v(ws[0], ws[1]);
        *(v8bf_t*)&sW[0][sr * 32 + sk + 8] = cvt8v(ws[2], ws[3]);
        if constexpr (sizeof(AT) == 2) async_wait0();
        __syncthreads();
    }

    int p = 0;
    for (int k0 = 0; k0 < K; k0 += 32, p ^= 1) {
        const bool nxt = (k0 + 32) < K;
        float4 a0{}, a1{}, a2{}, a3{}, w0{}, w1{}, w2{}, w3{};
        // ---- 1) issue global fetches for tile k0+32 (overlap with WMMA) ----
        if (nxt) {
            if constexpr (sizeof(AT) == 4) {
                const float4* as =
                    (const float4*)(A + (size_t)(m0 + sr) * K + k0 + 32 + sk);
                a0 = as[0]; a1 = as[1]; a2 = as[2]; a3 = as[3];
                __builtin_prefetch((const float*)as + 32, 0, 0);
            } else {
                const __bf16* as =
                    (const __bf16*)A + (size_t)(m0 + sr) * K + k0 + 32 + sk;
                const unsigned l = lds_addr(&sA[p ^ 1][sr * 32 + sk]);
                async_cp16(l, as);
                async_cp16(l + 16, as + 8);
                __builtin_prefetch(as + 32, 0, 0);
            }
            const float4* ws =
                (const float4*)(W + (size_t)(n0 + sr) * K + k0 + 32 + sk);
            w0 = ws[0]; w1 = ws[1]; w2 = ws[2]; w3 = ws[3];
        }

        // ---- 2) compute on buffer p ----
        v16bf_t af[2], bfr[4];
#pragma unroll
        for (int mf = 0; mf < 2; ++mf) {
            const __bf16* q = &sA[p][(wrow * 32 + mf * 16 + ln) * 32 + 8 * half];
            af[mf] = combine8(ld8(q), ld8(q + 16));
        }
#pragma unroll
        for (int nf = 0; nf < 4; ++nf) {
            const __bf16* q = &sW[p][(wcol * 64 + nf * 16 + ln) * 32 + 8 * half];
            bfr[nf] = combine8(ld8(q), ld8(q + 16));
        }
#pragma unroll
        for (int mf = 0; mf < 2; ++mf)
#pragma unroll
            for (int nf = 0; nf < 4; ++nf)
                acc[mf][nf] = wmma_bf16(af[mf], bfr[nf], acc[mf][nf]);

        // ---- 3) drain staged data into buffer p^1 ----
        if (nxt) {
            if constexpr (sizeof(AT) == 4) {
                *(v8bf_t*)&sA[p ^ 1][sr * 32 + sk]     = cvt8v(a0, a1);
                *(v8bf_t*)&sA[p ^ 1][sr * 32 + sk + 8] = cvt8v(a2, a3);
            }
            *(v8bf_t*)&sW[p ^ 1][sr * 32 + sk]     = cvt8v(w0, w1);
            *(v8bf_t*)&sW[p ^ 1][sr * 32 + sk + 8] = cvt8v(w2, w3);
            if constexpr (sizeof(AT) == 2) async_wait0();
        }
        __syncthreads();
    }

#pragma unroll
    for (int mf = 0; mf < 2; ++mf)
#pragma unroll
        for (int nf = 0; nf < 4; ++nf) {
            const int col = n0 + wcol * 64 + nf * 16 + ln;
            const float bv = bias[col];
#pragma unroll
            for (int r = 0; r < 8; ++r) {
                const int row = m0 + wrow * 32 + mf * 16 + r + 8 * half;
                C[(size_t)row * N + col] = (OT)(acc[mf][nf][r] + bv);
            }
        }
}

// ---------------------------------------------------------------------------
// Flash attention per (b,h). qkv (bf16) viewed as (B*H, 2048, 192):
//   Q = cols [0,64), K = [64,128), V = [128,192), row stride 192.
// Block: 128 query rows, 8 waves x 16 rows; 64-key tiles double-buffered:
// TDM fetches K[kt+1] and V[kt+1] loads fly while tile kt is computed.
// ctx (bf16) laid out (B,H,S,hd) contiguous == flat (B,S,D).
// ---------------------------------------------------------------------------
__global__ __launch_bounds__(256) void flash_attn_kernel(
    const __bf16* __restrict__ qkv, __bf16* __restrict__ ctx) {
    constexpr int   RS    = 192;
    constexpr float SCALE = 0.125f;  // 1/sqrt(64)

    __shared__ alignas(16) __bf16 sK[2][64 * 64];   // key tiles (TDM dest)
    __shared__ alignas(16) __bf16 sVt[2][64 * 72];  // V^T tiles, padded
    __shared__ alignas(16) __bf16 sP[8 * 16 * 72];  // per-wave P tiles, padded

    const int t    = threadIdx.x;
    const int lane = t & 31;
    const int wid  = t >> 5;
    const int half = lane >> 4;
    const int ln   = lane & 15;

    const int    qt   = blockIdx.x & 15;  // 16 query tiles of 128
    const int    bh   = blockIdx.x >> 4;  // (b*H + h), 0..31
    const size_t base = (size_t)bh * 2048 * RS;
    const int    q0   = qt * 128;

    // Q fragments (A-layout), loaded once from global bf16: lane row = ln
    const __bf16* qrow = qkv + base + (size_t)(q0 + wid * 16 + ln) * RS;
    v16bf_t qf[2];
#pragma unroll
    for (int kf = 0; kf < 2; ++kf)
        qf[kf] = combine8(ld8(qrow + kf * 32 + 8 * half),
                          ld8(qrow + kf * 32 + 16 + 8 * half));

    float mx[8], l[8];
    v8f_t o[4] = {};
#pragma unroll
    for (int r = 0; r < 8; ++r) { mx[r] = -3.0e38f; l[r] = 0.0f; }

    const int cr  = t >> 2;        // cooperative-load row (0..63)
    const int seg = (t & 3) * 16;  // 16-element segment

    // ---- prologue: fill buffer 0 with tile 0 ----
    if (wid == 0)
        tdm_load_tile_bf16(lds_addr(&sK[0][0]), qkv + base + 64, 64, 64, RS);
    {
        const __bf16* vsrc = qkv + base + (size_t)cr * RS + 128 + seg;
        const v8bf_t v0 = ld8(vsrc);
        const v8bf_t v1 = ld8(vsrc + 8);
#pragma unroll
        for (int c = 0; c < 8; ++c) sVt[0][(seg + c) * 72 + cr] = v0[c];
#pragma unroll
        for (int c = 0; c < 8; ++c) sVt[0][(seg + 8 + c) * 72 + cr] = v1[c];
    }
    if (wid == 0) tensor_wait0();
    __syncthreads();

    int p = 0;
    for (int kt = 0; kt < 32; ++kt, p ^= 1) {
        const bool nxt = (kt + 1) < 32;
        // ---- 1) issue fetches for tile kt+1 into buffer p^1 ----
        v8bf_t nv0{}, nv1{};
        if (nxt) {
            if (wid == 0)
                tdm_load_tile_bf16(lds_addr(&sK[p ^ 1][0]),
                                   qkv + base + (size_t)(kt + 1) * 64 * RS + 64,
                                   64, 64, RS);
            const __bf16* vsrc =
                qkv + base + (size_t)((kt + 1) * 64 + cr) * RS + 128 + seg;
            nv0 = ld8(vsrc);
            nv1 = ld8(vsrc + 8);
        }

        // ---- 2) scores S = Q * K^T (16 x 64 per wave) ----
        v8f_t s[4];
#pragma unroll
        for (int nf = 0; nf < 4; ++nf) {
            v8f_t c = {};
#pragma unroll
            for (int kf = 0; kf < 2; ++kf) {
                const __bf16* q = &sK[p][(nf * 16 + ln) * 64 + kf * 32 + 8 * half];
                c = wmma_bf16(qf[kf], combine8(ld8(q), ld8(q + 16)), c);
            }
            s[nf] = c;
        }

        // ---- online softmax (rows live across 16-lane halves) ----
        float mnew[8];
#pragma unroll
        for (int r = 0; r < 8; ++r) {
            mnew[r] = mx[r];
#pragma unroll
            for (int nf = 0; nf < 4; ++nf)
                mnew[r] = fmaxf(mnew[r], s[nf][r] * SCALE);
#pragma unroll
            for (int m = 1; m < 16; m <<= 1)
                mnew[r] = fmaxf(mnew[r], __shfl_xor(mnew[r], m, 16));
        }
        float alpha[8], rs[8];
#pragma unroll
        for (int r = 0; r < 8; ++r) {
            alpha[r] = __expf(mx[r] - mnew[r]);
            mx[r]    = mnew[r];
            rs[r]    = 0.0f;
        }
#pragma unroll
        for (int nf = 0; nf < 4; ++nf)
#pragma unroll
            for (int r = 0; r < 8; ++r) {
                const float pe = __expf(s[nf][r] * SCALE - mx[r]);
                s[nf][r] = pe;
                rs[r] += pe;
            }
#pragma unroll
        for (int r = 0; r < 8; ++r) {
#pragma unroll
            for (int m = 1; m < 16; m <<= 1)
                rs[r] += __shfl_xor(rs[r], m, 16);
            l[r] = l[r] * alpha[r] + rs[r];
        }
#pragma unroll
        for (int nf = 0; nf < 4; ++nf)
#pragma unroll
            for (int r = 0; r < 8; ++r)
                o[nf][r] *= alpha[r];

        // ---- C-layout -> A-layout for P via per-wave LDS tile ----
        __bf16* pw = &sP[wid * 16 * 72];
#pragma unroll
        for (int nf = 0; nf < 4; ++nf)
#pragma unroll
            for (int r = 0; r < 8; ++r)
                pw[(r + 8 * half) * 72 + nf * 16 + ln] = (__bf16)s[nf][r];

        v16bf_t pf[2];
#pragma unroll
        for (int kf = 0; kf < 2; ++kf) {
            const __bf16* q = &pw[ln * 72 + kf * 32 + 8 * half];
            pf[kf] = combine8(ld8(q), ld8(q + 16));
        }

        // ---- O += P * V ----
#pragma unroll
        for (int nf = 0; nf < 4; ++nf)
#pragma unroll
            for (int kf = 0; kf < 2; ++kf) {
                const __bf16* q = &sVt[p][(nf * 16 + ln) * 72 + kf * 32 + 8 * half];
                o[nf] = wmma_bf16(pf[kf], combine8(ld8(q), ld8(q + 16)), o[nf]);
            }

        // ---- 3) drain V[kt+1] transpose into buffer p^1 ----
        if (nxt) {
#pragma unroll
            for (int c = 0; c < 8; ++c) sVt[p ^ 1][(seg + c) * 72 + cr] = nv0[c];
#pragma unroll
            for (int c = 0; c < 8; ++c) sVt[p ^ 1][(seg + 8 + c) * 72 + cr] = nv1[c];
        }
        if (wid == 0) tensor_wait0();
        __syncthreads();
    }

    // ---- epilogue: O / l -> ctx (bf16) ----
#pragma unroll
    for (int nf = 0; nf < 4; ++nf)
#pragma unroll
        for (int r = 0; r < 8; ++r) {
            const int row = q0 + wid * 16 + r + 8 * half;
            ctx[((size_t)bh * 2048 + row) * 64 + nf * 16 + ln] =
                (__bf16)(o[nf][r] / l[r]);
        }
}

extern "C" void kernel_launch(void* const* d_in, const int* in_sizes, int n_in,
                              void* d_out, int out_size, void* d_ws, size_t ws_size,
                              hipStream_t stream) {
    (void)in_sizes; (void)n_in; (void)out_size; (void)ws_size;
    const float* x    = (const float*)d_in[0];  // (2,2048,1024)
    const float* Wqkv = (const float*)d_in[1];  // (3072,1024)
    const float* bqkv = (const float*)d_in[2];  // (3072,)
    const float* Wo   = (const float*)d_in[3];  // (1024,1024)
    const float* bo   = (const float*)d_in[4];  // (1024,)
    float* out = (float*)d_out;                 // (2,2048,1024)

    __bf16* qkv = (__bf16*)d_ws;                 // 4096 x 3072 bf16 (25.2 MB)
    __bf16* ctx = qkv + (size_t)4096 * 3072;     // 4096 x 1024 bf16 (8.4 MB)

    // qkv = x @ Wqkv^T + bqkv  (bf16 output)
    gemm_bias_kernel<float, __bf16>
        <<<dim3(3072 / 128, 4096 / 128), 256, 0, stream>>>(
            x, Wqkv, bqkv, qkv, 4096, 3072, 1024);
    // per-(b,h) flash attention (bf16 in/out)
    flash_attn_kernel<<<dim3(32 * 16), 256, 0, stream>>>(qkv, ctx);
    // out = ctx @ Wo^T + bo  (f32 output)
    gemm_bias_kernel<__bf16, float>
        <<<dim3(1024 / 128, 4096 / 128), 256, 0, stream>>>(
            ctx, Wo, bo, out, 4096, 1024, 1024);
}
